// UpsampleNet_82179904242034
// MI455X (gfx1250) — matrix operand order repaired and checked by reading, not compiled, not run on hardware
//
#include <hip/hip_runtime.h>

#define B_ 4
#define N_ 4096
#define C_ 512
#define R_ 4
#define NR_ (N_*R_)      // 16384 points per batch after augmentation
#define M_OUT_ (2*N_)    // 8192 FPS-selected points per batch

typedef __attribute__((ext_vector_type(16))) __bf16        v16bf;
typedef __attribute__((ext_vector_type(8)))  float         v8f;
typedef __attribute__((ext_vector_type(4)))  unsigned int  u32x4;

union Frag { v16bf v; u32x4 q[2]; unsigned short s[16]; };

__device__ __forceinline__ unsigned short f2bf(float f) {
    unsigned int u = __float_as_uint(f);
    u += 0x7FFFu + ((u >> 16) & 1u);   // round-to-nearest-even
    return (unsigned short)(u >> 16);
}
__device__ __forceinline__ float bf2f(unsigned short s) {
    return __uint_as_float(((unsigned)s) << 16);
}

// ---------------------------------------------------------------- bbox / stats
__global__ void bbox_kernel(const float* __restrict__ pcl, float* __restrict__ stats) {
    int b = blockIdx.x, t = threadIdx.x;
    const float* p = pcl + b * N_ * 3;
    float mn[3] = {1e30f, 1e30f, 1e30f}, mx[3] = {-1e30f, -1e30f, -1e30f};
    for (int i = t; i < N_; i += 256) {
        for (int d = 0; d < 3; ++d) {
            float v = p[i * 3 + d];
            mn[d] = fminf(mn[d], v);
            mx[d] = fmaxf(mx[d], v);
        }
    }
    __shared__ float smn[3][256], smx[3][256];
    for (int d = 0; d < 3; ++d) { smn[d][t] = mn[d]; smx[d][t] = mx[d]; }
    __syncthreads();
    for (int s = 128; s > 0; s >>= 1) {
        if (t < s)
            for (int d = 0; d < 3; ++d) {
                smn[d][t] = fminf(smn[d][t], smn[d][t + s]);
                smx[d][t] = fmaxf(smx[d][t], smx[d][t + s]);
            }
        __syncthreads();
    }
    if (t == 0) {
        float ex = fmaxf(smx[0][0] - smn[0][0],
                   fmaxf(smx[1][0] - smn[1][0], smx[2][0] - smn[2][0]));
        float scale = ex / (2.0f * 0.5f);   // BOX = 0.5
        for (int d = 0; d < 3; ++d) stats[b * 4 + d] = 0.5f * (smn[d][0] + smx[d][0]);
        stats[b * 4 + 3] = scale;
    }
}

// ------------------------------------------------- normalize + noise augment
__global__ void prep_kernel(const float* __restrict__ pcl, const float* __restrict__ noise,
                            const float* __restrict__ stats,
                            float* __restrict__ p_norm, float* __restrict__ pcl_noise) {
    int idx = blockIdx.x * 256 + threadIdx.x;   // over B_*NR_
    int b = idx / NR_, i = idx % NR_;
    int src = i & (N_ - 1);
    float inv = 1.0f / stats[b * 4 + 3];
    for (int d = 0; d < 3; ++d) {
        float p = (pcl[(b * N_ + src) * 3 + d] - stats[b * 4 + d]) * inv;
        if (i < N_) p_norm[(b * N_ + i) * 3 + d] = p;
        pcl_noise[(b * NR_ + i) * 3 + d] = p + noise[(b * NR_ + i) * 3 + d] * 0.01f;
    }
}

// -------- pack weights into bf16 WMMA B-fragment order (ISA 7.12.2, 16-bit B)
// lane L: n = ntile*16 + (L&15), kh = L>>4 ; element i: k = k0 + (i>>3)*16 + kh*8 + (i&7)
__global__ void pack_kernel(const float* __restrict__ W2, const float* __restrict__ Wg1,
                            unsigned short* __restrict__ W2p, unsigned short* __restrict__ Wg1p) {
    int idx = blockIdx.x * 256 + threadIdx.x;   // 512*512 packed slots
    int i = idx & 15, lane = (idx >> 4) & 31, kstep = (idx >> 9) & 15, ntile = idx >> 13;
    int n = ntile * 16 + (lane & 15);
    int k = kstep * 32 + (i >> 3) * 16 + (lane >> 4) * 8 + (i & 7);
    W2p[idx]  = f2bf(W2[k * C_ + n]);
    Wg1p[idx] = f2bf(Wg1[(k + 3) * C_ + n]);   // Wg1 rows 0..2 are xyz, 3..514 are feat
}

// ----------------------------------------- backbone: relu(p@W1+b1)@W2+b2 -> feat
// M-tile = 32 points/block: each B-fragment feeds 2 WMMAs (halves weight traffic)
__global__ void __launch_bounds__(256) backbone_kernel(
        const float* __restrict__ p_norm, const float* __restrict__ W1,
        const float* __restrict__ b1, const unsigned short* __restrict__ W2p,
        const float* __restrict__ b2, unsigned short* __restrict__ featb) {
    __shared__ __align__(16) unsigned short h1[32 * C_];  // 32 KB bf16 A tile
    __shared__ float xyz[96];
    int blk = blockIdx.x;
    int b = blk / (N_ / 32), tm = blk % (N_ / 32);
    int row0 = tm * 32;
    int t = threadIdx.x;
    if (t < 96) xyz[t] = p_norm[(b * N_ + row0 + t / 3) * 3 + (t % 3)];
    __syncthreads();
    {   // layer 1: K=3, pure VALU
        int m = t & 31, g = t >> 5;
        float x = xyz[m * 3 + 0], y = xyz[m * 3 + 1], z = xyz[m * 3 + 2];
        for (int c = g * 64; c < g * 64 + 64; ++c) {
            float a = b1[c] + x * W1[c] + y * W1[C_ + c] + z * W1[2 * C_ + c];
            h1[m * C_ + c] = f2bf(fmaxf(a, 0.0f));
        }
    }
    __syncthreads();
    int lane = t & 31, wave = t >> 5;
    int mL = lane & 15, kh = lane >> 4;
    v8f zero = {0.f, 0.f, 0.f, 0.f, 0.f, 0.f, 0.f, 0.f};
    v8f acc[2][4];
#pragma unroll
    for (int mt = 0; mt < 2; ++mt)
#pragma unroll
        for (int nt = 0; nt < 4; ++nt) acc[mt][nt] = zero;
    for (int k0 = 0; k0 < C_; k0 += 32) {
        Frag a0, a1;   // A layout: element i -> k = k0 + (i>>3)*16 + kh*8 + (i&7)
        a0.q[0] = *(const u32x4*)&h1[mL * C_ + k0 + kh * 8];
        a0.q[1] = *(const u32x4*)&h1[mL * C_ + k0 + 16 + kh * 8];
        a1.q[0] = *(const u32x4*)&h1[(16 + mL) * C_ + k0 + kh * 8];
        a1.q[1] = *(const u32x4*)&h1[(16 + mL) * C_ + k0 + 16 + kh * 8];
        Frag bf[4];
#pragma unroll
        for (int nt = 0; nt < 4; ++nt) {
            const u32x4* pb =
                (const u32x4*)&W2p[(((wave * 4 + nt) * 16 + (k0 >> 5)) * 32 + lane) * 16];
            bf[nt].q[0] = pb[0]; bf[nt].q[1] = pb[1];
        }
#pragma unroll
        for (int nt = 0; nt < 4; ++nt) {
            acc[0][nt] = __builtin_amdgcn_wmma_f32_16x16x32_bf16(
                false, a0.v, false, bf[nt].v, (short)0, acc[0][nt], false, false);
            acc[1][nt] = __builtin_amdgcn_wmma_f32_16x16x32_bf16(
                false, a1.v, false, bf[nt].v, (short)0, acc[1][nt], false, false);
        }
    }
#pragma unroll
    for (int nt = 0; nt < 4; ++nt) {
        int n = (wave * 4 + nt) * 16 + mL;
        float bias = b2[n];
#pragma unroll
        for (int mt = 0; mt < 2; ++mt)
#pragma unroll
            for (int r = 0; r < 8; ++r) {
                int mr = mt * 16 + r + 8 * kh;  // C/D: m = r + 8*(lane>>4), n = lane&15
                featb[(b * N_ + row0 + mr) * C_ + n] = f2bf(acc[mt][nt][r] + bias);
            }
    }
}

// -------- generator: h = relu([xyz;feat]@Wg1+bg1); disp = h@Wg2+bg2 (fused)
// M-tile = 32; A tile staged with CDNA5 async global->LDS DMA (ASYNCcnt path);
// LDS A buffer is reused for the relu output (bf16) after the K loop.
__global__ void __launch_bounds__(256) generator_kernel(
        const unsigned short* __restrict__ featb, const float* __restrict__ pcl_noise,
        const unsigned short* __restrict__ Wg1p, const float* __restrict__ Wg1,
        const float* __restrict__ bg1, const float* __restrict__ Wg2,
        const float* __restrict__ bg2, const float* __restrict__ stats,
        float* __restrict__ pred_noise, float* __restrict__ newp) {
    __shared__ __align__(16) unsigned short sbuf[32 * C_]; // 32 KB: A tile, then h
    __shared__ float xyzs[96];
    __shared__ float red[96];
    int blk = blockIdx.x;
    int b = blk / (NR_ / 32), ti = blk % (NR_ / 32);
    int i0 = ti * 32;
    int src0 = i0 & (N_ - 1);   // feat_rep rows are a contiguous 32-row block
    int t = threadIdx.x;
    {   // async bulk copy: 32 rows x 512 bf16 = 32 KB, 128 B per thread
        unsigned long long ga =
            (unsigned long long)(featb + ((size_t)b * N_ + src0) * C_) +
            (unsigned long long)(t * 128);
        unsigned la = (unsigned)(unsigned long long)(void*)&sbuf[0] + (unsigned)(t * 128);
#pragma unroll
        for (int j = 0; j < 8; ++j) {
            asm volatile("global_load_async_to_lds_b128 %0, %1, off"
                         :: "v"(la + (unsigned)(j * 16)),
                            "v"(ga + (unsigned long long)(j * 16))
                         : "memory");
        }
    }
    if (t < 96) { xyzs[t] = pcl_noise[(b * NR_ + i0 + t / 3) * 3 + (t % 3)]; red[t] = 0.0f; }
    asm volatile("s_wait_asynccnt 0x0" ::: "memory");
    __syncthreads();
    int lane = t & 31, wave = t >> 5;
    int mL = lane & 15, kh = lane >> 4;
    v8f zero = {0.f, 0.f, 0.f, 0.f, 0.f, 0.f, 0.f, 0.f};
    v8f acc[2][4];
#pragma unroll
    for (int mt = 0; mt < 2; ++mt)
#pragma unroll
        for (int nt = 0; nt < 4; ++nt) acc[mt][nt] = zero;
    for (int k0 = 0; k0 < C_; k0 += 32) {
        Frag a0, a1;
        a0.q[0] = *(const u32x4*)&sbuf[mL * C_ + k0 + kh * 8];
        a0.q[1] = *(const u32x4*)&sbuf[mL * C_ + k0 + 16 + kh * 8];
        a1.q[0] = *(const u32x4*)&sbuf[(16 + mL) * C_ + k0 + kh * 8];
        a1.q[1] = *(const u32x4*)&sbuf[(16 + mL) * C_ + k0 + 16 + kh * 8];
        Frag bf[4];
#pragma unroll
        for (int nt = 0; nt < 4; ++nt) {
            const u32x4* pb =
                (const u32x4*)&Wg1p[(((wave * 4 + nt) * 16 + (k0 >> 5)) * 32 + lane) * 16];
            bf[nt].q[0] = pb[0]; bf[nt].q[1] = pb[1];
        }
#pragma unroll
        for (int nt = 0; nt < 4; ++nt) {
            acc[0][nt] = __builtin_amdgcn_wmma_f32_16x16x32_bf16(
                false, a0.v, false, bf[nt].v, (short)0, acc[0][nt], false, false);
            acc[1][nt] = __builtin_amdgcn_wmma_f32_16x16x32_bf16(
                false, a1.v, false, bf[nt].v, (short)0, acc[1][nt], false, false);
        }
    }
    __syncthreads();   // all waves done reading A tile; sbuf is now reused for h
#pragma unroll
    for (int nt = 0; nt < 4; ++nt) {   // += xyz part (Wg1 rows 0..2) + bias, relu
        int n = (wave * 4 + nt) * 16 + mL;
        float w0 = Wg1[n], w1 = Wg1[C_ + n], w2 = Wg1[2 * C_ + n];
        float bias = bg1[n];
#pragma unroll
        for (int mt = 0; mt < 2; ++mt)
#pragma unroll
            for (int r = 0; r < 8; ++r) {
                int mr = mt * 16 + r + 8 * kh;
                float v = acc[mt][nt][r] + bias + xyzs[mr * 3] * w0
                        + xyzs[mr * 3 + 1] * w1 + xyzs[mr * 3 + 2] * w2;
                sbuf[mr * C_ + n] = f2bf(fmaxf(v, 0.0f));
            }
    }
    __syncthreads();
    {   // 512 -> 3 projection with LDS float-atomic reduction
        int m = t & 31, g = t >> 5;
        float p0 = 0.f, p1 = 0.f, p2 = 0.f;
        for (int c = g * 64; c < g * 64 + 64; ++c) {
            float hv = bf2f(sbuf[m * C_ + c]);
            p0 += hv * Wg2[c * 3 + 0];
            p1 += hv * Wg2[c * 3 + 1];
            p2 += hv * Wg2[c * 3 + 2];
        }
        atomicAdd(&red[m * 3 + 0], p0);
        atomicAdd(&red[m * 3 + 1], p1);
        atomicAdd(&red[m * 3 + 2], p2);
    }
    __syncthreads();
    if (t < 96) {
        int m = t / 3, j = t % 3;
        float disp = red[t] + bg2[j];
        float pn = xyzs[t];
        pred_noise[(b * NR_ + i0 + m) * 3 + j] = disp;
        newp[(b * NR_ + i0 + m) * 3 + j] = (pn + disp) * stats[b * 4 + 3] + stats[b * 4 + j];
    }
}

// --------------------------------------- farthest point sampling, 1 block/batch
__global__ void __launch_bounds__(256) fps_kernel(const float* __restrict__ newp,
                                                  float* __restrict__ out) {
    int b = blockIdx.x, t = threadIdx.x;
    const float* pts = newp + (size_t)b * NR_ * 3;
    float* ob = out + (size_t)b * M_OUT_ * 3;
    float dist[NR_ / 256];
#pragma unroll
    for (int k = 0; k < NR_ / 256; ++k) dist[k] = 3.4e38f;
    __shared__ float sl[3];
    __shared__ float rv[256];
    __shared__ int ri[256];
    int sel = 0;
    for (int it = 0; it < M_OUT_; ++it) {
        if (t == 0) {
            float px = pts[sel * 3], py = pts[sel * 3 + 1], pz = pts[sel * 3 + 2];
            ob[it * 3] = px; ob[it * 3 + 1] = py; ob[it * 3 + 2] = pz;
            sl[0] = px; sl[1] = py; sl[2] = pz;
        }
        __syncthreads();
        float lx = sl[0], ly = sl[1], lz = sl[2];
        float best = -1.0f; int bi = 0;
#pragma unroll
        for (int k = 0; k < NR_ / 256; ++k) {
            int idx = k * 256 + t;
            float dx = pts[idx * 3] - lx, dy = pts[idx * 3 + 1] - ly, dz = pts[idx * 3 + 2] - lz;
            float d = dx * dx + dy * dy + dz * dz;
            float dm = fminf(dist[k], d);
            dist[k] = dm;
            if (dm > best) { best = dm; bi = idx; }
        }
        rv[t] = best; ri[t] = bi;
        __syncthreads();
        for (int s = 128; s > 0; s >>= 1) {
            if (t < s && rv[t + s] > rv[t]) { rv[t] = rv[t + s]; ri[t] = ri[t + s]; }
            __syncthreads();
        }
        sel = ri[0];
        __syncthreads();
    }
}

extern "C" void kernel_launch(void* const* d_in, const int* in_sizes, int n_in,
                              void* d_out, int out_size, void* d_ws, size_t ws_size,
                              hipStream_t stream) {
    (void)in_sizes; (void)n_in; (void)out_size; (void)ws_size;
    const float* pcl   = (const float*)d_in[0];
    const float* noise = (const float*)d_in[1];
    const float* W1    = (const float*)d_in[2];
    const float* b1    = (const float*)d_in[3];
    const float* W2    = (const float*)d_in[4];
    const float* b2    = (const float*)d_in[5];
    const float* Wg1   = (const float*)d_in[6];
    const float* bg1   = (const float*)d_in[7];
    const float* Wg2   = (const float*)d_in[8];
    const float* bg2   = (const float*)d_in[9];

    char* ws = (char*)d_ws;
    float*          stats     = (float*)(ws + 0);          //  64 B
    float*          p_norm    = (float*)(ws + 256);        // 196608 B
    float*          pcl_noise = (float*)(ws + 196864);     // 786432 B
    unsigned short* featb     = (unsigned short*)(ws + 983296);   // 16.8 MB bf16
    float*          newp      = (float*)(ws + 17760512);   // 786432 B
    unsigned short* W2p       = (unsigned short*)(ws + 18546944); // 512 KB
    unsigned short* Wg1p      = (unsigned short*)(ws + 19071232); // 512 KB

    float* new_points = (float*)d_out;                        // (B, 8192, 3)
    float* pred_noise = new_points + (size_t)B_ * M_OUT_ * 3; // (B, 16384, 3)

    bbox_kernel<<<B_, 256, 0, stream>>>(pcl, stats);
    prep_kernel<<<(B_ * NR_) / 256, 256, 0, stream>>>(pcl, noise, stats, p_norm, pcl_noise);
    pack_kernel<<<(C_ * C_) / 256, 256, 0, stream>>>(W2, Wg1, W2p, Wg1p);
    backbone_kernel<<<B_ * N_ / 32, 256, 0, stream>>>(p_norm, W1, b1, W2p, b2, featb);
    generator_kernel<<<B_ * NR_ / 32, 256, 0, stream>>>(featb, pcl_noise, Wg1p, Wg1,
                                                        bg1, Wg2, bg2, stats,
                                                        pred_noise, newp);
    fps_kernel<<<B_, 256, 0, stream>>>(newp, new_points);
}
